// CapsuleNetwork_79620103733719
// MI455X (gfx1250) — compile-verified, matrix operand order checked by reading, not run on hardware
//
#include <hip/hip_runtime.h>
#include <cstddef>
#include <cstdint>

#define B_  256
#define S_  256
#define H_  768
#define DA_ 350
#define R_  30
#define C_  20
#define P_  16
#define NUM_ITERS 3
#define DA_PAD 352     // 11 k-steps of 32; cols 350/351 zeroed on the ws2 side
#define NT1 22         // 16-wide n-tiles covering DA_PAD

typedef __bf16 bf16;
typedef bf16  v8bf  __attribute__((ext_vector_type(8)));
typedef bf16  v16bf __attribute__((ext_vector_type(16)));
typedef float v8f   __attribute__((ext_vector_type(8)));

// D(16x16,f32) = A(16x32,bf16) * B(32x16,bf16) + C
// 16-bit A layout: lane l<16 = row M=l, elems 0-7 -> K=k0..k0+7, 8-15 -> K=k0+16..+23
//                  lane l>=16 = row M=l-16, elems 0-7 -> K=k0+8..+15, 8-15 -> K=k0+24..+31
// B mirrors with N=lane%16. C/D: vgpr i -> (M=i+8*half, N=lane%16).
__device__ __forceinline__ v8f wmma_bf16(v16bf a, v16bf b, v8f c) {
  return __builtin_amdgcn_wmma_f32_16x16x32_bf16(false, a, false, b, (short)0, c,
                                                 false, false);
}

__device__ __forceinline__ v16bf join16(v8bf lo, v8bf hi) {
  return __builtin_shufflevector(lo, hi, 0, 1, 2, 3, 4, 5, 6, 7,
                                 8, 9, 10, 11, 12, 13, 14, 15);
}

// convert two contiguous 8-float chunks to the 16-element bf16 A/B fragment
__device__ __forceinline__ v16bf cvt16(const float* __restrict__ p0,
                                       const float* __restrict__ p1) {
  const float4 a0 = *(const float4*)(p0);
  const float4 a1 = *(const float4*)(p0 + 4);
  const float4 a2 = *(const float4*)(p1);
  const float4 a3 = *(const float4*)(p1 + 4);
  v16bf r;
  r[0] = (bf16)a0.x; r[1] = (bf16)a0.y; r[2]  = (bf16)a0.z; r[3]  = (bf16)a0.w;
  r[4] = (bf16)a1.x; r[5] = (bf16)a1.y; r[6]  = (bf16)a1.z; r[7]  = (bf16)a1.w;
  r[8] = (bf16)a2.x; r[9] = (bf16)a2.y; r[10] = (bf16)a2.z; r[11] = (bf16)a2.w;
  r[12] = (bf16)a3.x; r[13] = (bf16)a3.y; r[14] = (bf16)a3.z; r[15] = (bf16)a3.w;
  return r;
}

// ---------------------------------------------------------------------------
// Kernel 0a: convert ws1 -> bf16 [DA,H]; ws2 -> bf16 [R, DA_PAD] zero padded.
// ---------------------------------------------------------------------------
__global__ __launch_bounds__(256) void k_convert(const float* __restrict__ ws1,
                                                 const float* __restrict__ ws2,
                                                 bf16* __restrict__ ws1b,
                                                 bf16* __restrict__ ws2b) {
  const int gid = blockIdx.x * 256 + threadIdx.x;
  const int stride = gridDim.x * 256;
  for (int i = gid; i < DA_ * H_; i += stride) ws1b[i] = (bf16)ws1[i];
  for (int i = gid; i < R_ * DA_PAD; i += stride) {
    const int r = i / DA_PAD, a = i % DA_PAD;
    ws2b[i] = (a < DA_) ? (bf16)ws2[r * DA_ + a] : (bf16)0.0f;
  }
}

// ---------------------------------------------------------------------------
// Kernel 0b: WT[r][d][h] = (bf16)W[r][h][d] via LDS 32x32 tile transpose.
// grid: R * (H/32) * (320/32) = 7200 blocks of (32,8).
// ---------------------------------------------------------------------------
__global__ __launch_bounds__(256) void k_twt(const float* __restrict__ W,
                                             bf16* __restrict__ WT) {
  __shared__ float t[32][33];
  const int bid = blockIdx.x;
  const int r   = bid / ((H_ / 32) * (C_ * P_ / 32));
  const int rem = bid % ((H_ / 32) * (C_ * P_ / 32));
  const int h0  = (rem / (C_ * P_ / 32)) * 32;
  const int n0  = (rem % (C_ * P_ / 32)) * 32;
  const int tx = threadIdx.x, ty = threadIdx.y;
#pragma unroll
  for (int i = 0; i < 4; ++i)
    t[ty + i * 8][tx] = W[((size_t)r * H_ + h0 + ty + i * 8) * (C_ * P_) + n0 + tx];
  __syncthreads();
#pragma unroll
  for (int i = 0; i < 4; ++i)
    WT[((size_t)r * (C_ * P_) + n0 + ty + i * 8) * H_ + h0 + tx] =
        (bf16)t[tx][ty + i * 8];
}

// ---------------------------------------------------------------------------
// Kernel 1: pa = tanh(output @ ws1^T) in LDS (bf16), then attention logits
//           via a second WMMA against ws2b. grid: B*(S/16), 8 waves.
// ---------------------------------------------------------------------------
__global__ __launch_bounds__(256) void k_preatt(const float* __restrict__ out3,
                                                const bf16* __restrict__ ws1b,
                                                const bf16* __restrict__ ws2b,
                                                float* __restrict__ attn) {
  __shared__ bf16 pa[16 * DA_PAD];
  const int tid  = threadIdx.x;
  const int wave = tid >> 5;
  const int lane = tid & 31;
  const int half = lane >> 4;
  const int lm   = lane & 15;
  const int b    = blockIdx.x >> 4;
  const int s0   = (blockIdx.x & 15) * 16;
  const int o1 = 8 * half, o2 = 16 + 8 * half;

  const int t0 = wave, t1 = wave + 8, t2 = wave + 16;
  const bool has2 = (t2 < NT1);
  const float* Arow = out3 + ((size_t)b * S_ + (s0 + lm)) * H_;
  int n0 = t0 * 16 + lm; if (n0 > DA_ - 1) n0 = DA_ - 1;
  int n1 = t1 * 16 + lm; if (n1 > DA_ - 1) n1 = DA_ - 1;
  int n2 = has2 ? t2 * 16 + lm : n0; if (n2 > DA_ - 1) n2 = DA_ - 1;
  const bf16* B0 = ws1b + (size_t)n0 * H_;
  const bf16* B1 = ws1b + (size_t)n1 * H_;
  const bf16* B2 = ws1b + (size_t)n2 * H_;

  v8f c0 = {}; v8f c1 = {}; v8f c2 = {};
  for (int k = 0; k < H_; k += 32) {
    const v16bf a  = cvt16(Arow + k + o1, Arow + k + o2);
    const v16bf b0 = join16(*(const v8bf*)(B0 + k + o1), *(const v8bf*)(B0 + k + o2));
    const v16bf b1 = join16(*(const v8bf*)(B1 + k + o1), *(const v8bf*)(B1 + k + o2));
    const v16bf b2 = join16(*(const v8bf*)(B2 + k + o1), *(const v8bf*)(B2 + k + o2));
    c0 = wmma_bf16(a, b0, c0);
    c1 = wmma_bf16(a, b1, c1);
    c2 = wmma_bf16(a, b2, c2);   // waves 6,7: dummy tile, never stored
  }

#pragma unroll
  for (int i = 0; i < 8; ++i) {
    const int row = i + 8 * half;
    pa[row * DA_PAD + t0 * 16 + lm] = (bf16)tanhf(c0[i]);
    pa[row * DA_PAD + t1 * 16 + lm] = (bf16)tanhf(c1[i]);
    if (has2) pa[row * DA_PAD + t2 * 16 + lm] = (bf16)tanhf(c2[i]);
  }
  __syncthreads();

  // phase 2: logits[16 s-rows x 32 r-cols] = pa (16x352) @ ws2b^T; K-pad is 0.
  if (wave < 2) {
    const int r = wave * 16 + lm;
    const int rc = r > R_ - 1 ? R_ - 1 : r;
    const bf16* Bw = ws2b + (size_t)rc * DA_PAD;
    const bf16* Al = pa + lm * DA_PAD;
    v8f acc = {};
    for (int k = 0; k < DA_PAD; k += 32) {
      const v16bf a  = join16(*(const v8bf*)(Al + k + o1), *(const v8bf*)(Al + k + o2));
      const v16bf bb = join16(*(const v8bf*)(Bw + k + o1), *(const v8bf*)(Bw + k + o2));
      acc = wmma_bf16(a, bb, acc);
    }
    if (r < R_) {
#pragma unroll
      for (int i = 0; i < 8; ++i)
        attn[((size_t)b * R_ + r) * S_ + s0 + i + 8 * half] = acc[i];
    }
  }
}

// ---------------------------------------------------------------------------
// Kernel 2: softmax over S in place + bf16 copy. One block per (b,r).
// ---------------------------------------------------------------------------
__global__ __launch_bounds__(256) void k_softmax(float* __restrict__ attn,
                                                 bf16* __restrict__ attb) {
  __shared__ float red[256];
  const int tid = threadIdx.x;
  float* row = attn + (size_t)blockIdx.x * S_;
  const float x = row[tid];
  red[tid] = x; __syncthreads();
  for (int s = 128; s > 0; s >>= 1) {
    if (tid < s) red[tid] = fmaxf(red[tid], red[tid + s]);
    __syncthreads();
  }
  const float m = red[0]; __syncthreads();
  const float e = __expf(x - m);
  red[tid] = e; __syncthreads();
  for (int s = 128; s > 0; s >>= 1) {
    if (tid < s) red[tid] += red[tid + s];
    __syncthreads();
  }
  const float v = e * (1.0f / red[0]);
  row[tid] = v;
  attb[(size_t)blockIdx.x * S_ + tid] = (bf16)v;
}

// ---------------------------------------------------------------------------
// Kernel 3: sv[b] = attention[b] (RxS) @ output[b] (SxH), bf16 in / bf16 out.
// grid: B*6 blocks of 8 waves; wave -> one 16-wide n-tile (48), M padded to 32.
// ---------------------------------------------------------------------------
__global__ __launch_bounds__(256) void k_semantic(const bf16* __restrict__ attb,
                                                  const float* __restrict__ out3,
                                                  bf16* __restrict__ svb) {
  const int tid  = threadIdx.x;
  const int wave = tid >> 5;
  const int lane = tid & 31;
  const int half = lane >> 4;
  const int lm   = lane & 15;
  const int b     = blockIdx.x / 6;
  const int ntile = (blockIdx.x % 6) * 8 + wave;
  const int n     = ntile * 16 + lm;
  const int o1 = 8 * half, o2 = 16 + 8 * half;

  const int m0 = lm;
  int m1 = 16 + lm; if (m1 > R_ - 1) m1 = R_ - 1;   // rows 30,31 clamped, unstored
  const bf16* A0 = attb + ((size_t)b * R_ + m0) * S_;
  const bf16* A1 = attb + ((size_t)b * R_ + m1) * S_;
  const float* Bp = out3 + (size_t)b * S_ * H_ + n;

  v8f c0 = {}; v8f c1 = {};
  for (int k = 0; k < S_; k += 32) {
    const v16bf a0 = join16(*(const v8bf*)(A0 + k + o1), *(const v8bf*)(A0 + k + o2));
    const v16bf a1 = join16(*(const v8bf*)(A1 + k + o1), *(const v8bf*)(A1 + k + o2));
    v16bf bb;
#pragma unroll
    for (int j = 0; j < 8; ++j) {
      bb[j]     = (bf16)Bp[(size_t)(k + o1 + j) * H_];
      bb[j + 8] = (bf16)Bp[(size_t)(k + o2 + j) * H_];
    }
    c0 = wmma_bf16(a0, bb, c0);
    c1 = wmma_bf16(a1, bb, c1);
  }
#pragma unroll
  for (int i = 0; i < 8; ++i) {
    const int row0 = i + 8 * half;
    svb[((size_t)b * R_ + row0) * H_ + n] = (bf16)c0[i];
    const int row1 = 16 + i + 8 * half;
    if (row1 < R_) svb[((size_t)b * R_ + row1) * H_ + n] = (bf16)c1[i];
  }
}

// ---------------------------------------------------------------------------
// Kernel 4: prediction[:,r,:] = sv[:,r,:] (BxH) @ WT[r]^T, f32 result.
// 9600 waves (R * 16 mtiles * 20 ntiles) -> 1200 blocks of 8 waves.
// ---------------------------------------------------------------------------
__global__ __launch_bounds__(256) void k_predict(const bf16* __restrict__ svb,
                                                 const bf16* __restrict__ WT,
                                                 float* __restrict__ pred) {
  const int tid  = threadIdx.x;
  const int wave = tid >> 5;
  const int lane = tid & 31;
  const int half = lane >> 4;
  const int lm   = lane & 15;
  const int wid   = blockIdx.x * 8 + wave;
  const int r     = wid / (16 * 20);
  const int rem   = wid % (16 * 20);
  const int mtile = rem / 20;
  const int ntile = rem % 20;
  const int m = mtile * 16 + lm;
  const int n = ntile * 16 + lm;
  const int o1 = 8 * half, o2 = 16 + 8 * half;

  const bf16* A  = svb + ((size_t)m * R_ + r) * H_;
  const bf16* Bp = WT + ((size_t)r * (C_ * P_) + n) * H_;

  v8f c = {};
  for (int k = 0; k < H_; k += 32) {
    const v16bf a  = join16(*(const v8bf*)(A + k + o1), *(const v8bf*)(A + k + o2));
    const v16bf bb = join16(*(const v8bf*)(Bp + k + o1), *(const v8bf*)(Bp + k + o2));
    c = wmma_bf16(a, bb, c);
  }
#pragma unroll
  for (int i = 0; i < 8; ++i) {
    const int bi = mtile * 16 + i + 8 * half;
    pred[((size_t)bi * R_ + r) * (C_ * P_) + n] = c[i];
  }
}

// ---------------------------------------------------------------------------
// Kernel 5: dynamic routing entirely in LDS (f32). One 320-thread block per b.
// ---------------------------------------------------------------------------
__global__ __launch_bounds__(320) void k_route(const float* __restrict__ pred,
                                               float* __restrict__ cls,
                                               float* __restrict__ routes_out) {
  __shared__ float sp[R_ * C_ * P_];
  __shared__ float slog[R_ * C_];
  __shared__ float srt[R_ * C_];
  __shared__ float sact[C_ * P_];
  __shared__ float spre[C_ * P_];
  const int tid = threadIdx.x;
  const int b = blockIdx.x;
  const float* pb = pred + (size_t)b * R_ * C_ * P_;
  for (int i = tid; i < R_ * C_ * P_; i += 320) sp[i] = pb[i];
  for (int i = tid; i < R_ * C_; i += 320) slog[i] = 0.f;
  __syncthreads();

  const int c = tid >> 4;

  for (int it = 0; it < NUM_ITERS; ++it) {
    if (tid < R_) {
      const int r = tid;
      float mx = -1e30f;
      for (int cc = 0; cc < C_; ++cc) mx = fmaxf(mx, slog[r * C_ + cc]);
      float sum = 0.f;
      for (int cc = 0; cc < C_; ++cc) sum += __expf(slog[r * C_ + cc] - mx);
      const float inv = 1.f / sum;
      for (int cc = 0; cc < C_; ++cc)
        srt[r * C_ + cc] = __expf(slog[r * C_ + cc] - mx) * inv;
    }
    __syncthreads();
    float s = 0.f;
    for (int r = 0; r < R_; ++r) s += srt[r * C_ + c] * sp[r * C_ * P_ + tid];
    spre[tid] = s;
    __syncthreads();
    float n2 = 0.f;
    for (int pp = 0; pp < P_; ++pp) { const float v = spre[c * P_ + pp]; n2 += v * v; }
    const float scale = sqrtf(n2) / (0.5f + n2);   // x/n * n^2/(0.5+n^2)
    sact[tid] = s * scale;
    __syncthreads();
    for (int idx = tid; idx < R_ * C_; idx += 320) {
      const int r = idx / C_, cc = idx % C_;
      float d = 0.f;
      for (int pp = 0; pp < P_; ++pp)
        d += sp[r * C_ * P_ + cc * P_ + pp] * sact[cc * P_ + pp];
      slog[idx] += d;
    }
    __syncthreads();
  }

  if (tid < C_) {
    float n2 = 0.f;
    for (int pp = 0; pp < P_; ++pp) { const float v = sact[tid * P_ + pp]; n2 += v * v; }
    cls[(size_t)b * C_ + tid] = sqrtf(n2);
  }
  for (int i = tid; i < R_ * C_; i += 320)
    routes_out[(size_t)b * R_ * C_ + i] = srt[i];
}

// ---------------------------------------------------------------------------
extern "C" void kernel_launch(void* const* d_in, const int* in_sizes, int n_in,
                              void* d_out, int out_size, void* d_ws, size_t ws_size,
                              hipStream_t stream) {
  (void)in_sizes; (void)n_in; (void)out_size; (void)ws_size;
  const float* out3 = (const float*)d_in[0];   // [B,S,H]
  const float* ws1  = (const float*)d_in[1];   // [DA,H]
  const float* ws2  = (const float*)d_in[2];   // [R,DA]
  const float* W    = (const float*)d_in[3];   // [R,H,C*P]

  float* attn   = (float*)d_out;                         // [B,R,S]
  float* cls    = attn + (size_t)B_ * R_ * S_;           // [B,C]
  float* pred   = cls + (size_t)B_ * C_;                 // [B,R,C*P]
  float* routes = pred + (size_t)B_ * R_ * C_ * P_;      // [B,R,C]

  uint8_t* wp = (uint8_t*)d_ws;
  auto alloc = [&](size_t bytes) {
    uint8_t* p = wp;
    wp += (bytes + 255) & ~(size_t)255;
    return p;
  };
  bf16* ws1b = (bf16*)alloc((size_t)DA_ * H_ * sizeof(bf16));
  bf16* ws2b = (bf16*)alloc((size_t)R_ * DA_PAD * sizeof(bf16));
  bf16* WT   = (bf16*)alloc((size_t)R_ * C_ * P_ * H_ * sizeof(bf16));
  bf16* attb = (bf16*)alloc((size_t)B_ * R_ * S_ * sizeof(bf16));
  bf16* svb  = (bf16*)alloc((size_t)B_ * R_ * H_ * sizeof(bf16));

  k_convert <<<512, 256, 0, stream>>>(ws1, ws2, ws1b, ws2b);
  k_twt     <<<R_ * (H_ / 32) * (C_ * P_ / 32), dim3(32, 8), 0, stream>>>(W, WT);
  k_preatt  <<<B_ * (S_ / 16), 256, 0, stream>>>(out3, ws1b, ws2b, attn);
  k_softmax <<<B_ * R_, 256, 0, stream>>>(attn, attb);
  k_semantic<<<B_ * 6, 256, 0, stream>>>(attb, out3, svb);
  k_predict <<<(R_ * 16 * 20) / 8, 256, 0, stream>>>(svb, WT, pred);
  k_route   <<<B_, 320, 0, stream>>>(pred, cls, routes);
}
